// MiniStatefulDecoder_18554258718785
// MI455X (gfx1250) — matrix operand order repaired and checked by reading, not compiled, not run on hardware
//
#include <hip/hip_runtime.h>
#include <hip/hip_bf16.h>
#include <math.h>

// ---------------- model constants ----------------
#define H      1024
#define NQ     16
#define NKV    8
#define REP    2
#define HD     128
#define INTER  4096
#define VOCAB  1000
#define SEQ    16384
#define EPSV   1e-6f
#define SCALEV 0.08838834764831845f   // 1/sqrt(128)
#define SPLIT  64                     // seq splits for attn*V

typedef _Float16 v16h __attribute__((ext_vector_type(16)));
typedef _Float16 v8h  __attribute__((ext_vector_type(8)));
typedef float    v8f  __attribute__((ext_vector_type(8)));

// ---------------- helpers ----------------
__device__ __forceinline__ float wave_reduce_sum(float v) {
#pragma unroll
  for (int off = 16; off > 0; off >>= 1) v += __shfl_xor(v, off, 32);
  return v;
}

// ---------------- RMSNorm (n = 1024, one block of 256) ----------------
__global__ void rmsnorm_kernel(const float* __restrict__ x,
                               const float* __restrict__ w,
                               float* __restrict__ y) {
  __shared__ float red[256];
  int tid = threadIdx.x;
  float ss = 0.f;
  for (int i = tid; i < H; i += 256) { float v = x[i]; ss += v * v; }
  red[tid] = ss; __syncthreads();
  for (int k = 128; k > 0; k >>= 1) { if (tid < k) red[tid] += red[tid + k]; __syncthreads(); }
  float scale = rsqrtf(red[0] / (float)H + EPSV);
  for (int i = tid; i < H; i += 256) y[i] = x[i] * scale * w[i];
}

// ---------------- GEMV: y[r] = dot(W[r,:], x) (+res) ----------------
// one wave32 per row, 8 rows per 256-thread block, float4 streaming
__global__ void gemv_kernel(const float* __restrict__ W,
                            const float* __restrict__ x,
                            float* __restrict__ y,
                            const float* __restrict__ res,
                            int ncols, int nrows) {
  int row  = blockIdx.x * 8 + (threadIdx.x >> 5);
  int lane = threadIdx.x & 31;
  if (row >= nrows) return;
  const float4* w4 = (const float4*)(W + (size_t)row * ncols);
  const float4* x4 = (const float4*)x;
  int n4 = ncols >> 2;
  float acc = 0.f;
  for (int i = lane; i < n4; i += 32) {
    float4 a = w4[i], b = x4[i];
    acc = fmaf(a.x, b.x, acc); acc = fmaf(a.y, b.y, acc);
    acc = fmaf(a.z, b.z, acc); acc = fmaf(a.w, b.w, acc);
  }
  acc = wave_reduce_sum(acc);
  if (lane == 0) y[row] = acc + (res ? res[row] : 0.f);
}

// ---------------- fused gate/up GEMV + SiLU ----------------
__global__ void gateup_kernel(const float* __restrict__ Wg,
                              const float* __restrict__ Wu,
                              const float* __restrict__ x,
                              float* __restrict__ act,
                              int ncols, int nrows) {
  int row  = blockIdx.x * 8 + (threadIdx.x >> 5);
  int lane = threadIdx.x & 31;
  if (row >= nrows) return;
  const float4* g4 = (const float4*)(Wg + (size_t)row * ncols);
  const float4* u4 = (const float4*)(Wu + (size_t)row * ncols);
  const float4* x4 = (const float4*)x;
  int n4 = ncols >> 2;
  float ag = 0.f, au = 0.f;
  for (int i = lane; i < n4; i += 32) {
    float4 a = g4[i], b = u4[i], c = x4[i];
    ag = fmaf(a.x, c.x, ag); ag = fmaf(a.y, c.y, ag);
    ag = fmaf(a.z, c.z, ag); ag = fmaf(a.w, c.w, ag);
    au = fmaf(b.x, c.x, au); au = fmaf(b.y, c.y, au);
    au = fmaf(b.z, c.z, au); au = fmaf(b.w, c.w, au);
  }
  ag = wave_reduce_sum(ag);
  au = wave_reduce_sum(au);
  if (lane == 0) act[row] = (ag / (1.f + __expf(-ag))) * au;   // silu(g)*u
}

// ---------------- RoPE + fp16 staging ----------------
// Builds:
//  qB[NKV][16][HD] fp16 : per-kv-head zero-padded B operand for WMMA
//                         (slot 0/1 = RoPE'd q heads 2kv/2kv+1, slots 2..15 = 0)
//  k16new[NKV*HD]       : RoPE'd new k (fp16)
//  v16new[NKV*HD]       : new v (fp16)
// idx 0 .. NKV*16*HD-1          -> qB
// next NKV*HD                   -> k16new
// next NKV*HD                   -> v16new
__global__ void rope_stage_kernel(const float* __restrict__ q,
                                  const float* __restrict__ k,
                                  const float* __restrict__ v,
                                  const float* __restrict__ cosv,
                                  const float* __restrict__ sinv,
                                  _Float16* __restrict__ qB,
                                  _Float16* __restrict__ k16,
                                  _Float16* __restrict__ v16) {
  const int NQB = NKV * 16 * HD;
  int idx = blockIdx.x * 256 + threadIdx.x;
  if (idx < NQB) {
    int d    = idx & (HD - 1);
    int slot = (idx >> 7) & 15;
    int kv   = idx >> 11;
    _Float16 out = (_Float16)0.f;
    if (slot < REP) {
      int base = (kv * REP + slot) * HD;
      float rot = (d < HD / 2) ? -q[base + d + HD / 2] : q[base + d - HD / 2];
      out = (_Float16)(q[base + d] * cosv[d] + rot * sinv[d]);
    }
    qB[idx] = out;
  } else if (idx < NQB + NKV * HD) {
    int j = idx - NQB; int d = j & (HD - 1); int base = j - d;
    float rot = (d < HD / 2) ? -k[base + d + HD / 2] : k[base + d - HD / 2];
    k16[j] = (_Float16)(k[j] * cosv[d] + rot * sinv[d]);
  } else if (idx < NQB + 2 * NKV * HD) {
    int j = idx - NQB - NKV * HD;
    v16[j] = (_Float16)v[j];
  }
}

// ---------------- attention scores via v_wmma_f32_16x16x32_f16 ----------------
// grid (SEQ/128, NKV); 256 threads = 8 waves; wave -> 16 positions.
// D[16x16] = A(K_tile 16pos x 32chunk) x B(Q^T 32 x 16; cols 0..1 = q heads of kv head)
// B fragments are loaded branch-free from the zero-padded qB staging buffer and
// hoisted out of the K-streaming loop, so the loop body is pure b128 loads + WMMA.
__global__ void attn_scores_kernel(const _Float16* __restrict__ kc,
                                   const _Float16* __restrict__ knew,
                                   const _Float16* __restrict__ qB,
                                   const float* __restrict__ mask,
                                   float* __restrict__ scores) {
  int wave = threadIdx.x >> 5;
  int lane = threadIdx.x & 31;
  int col  = lane & 15;        // B/D: lane = N column
  int half = lane >> 4;
  int kv   = blockIdx.y;
  int pos_base = blockIdx.x * 128 + wave * 16;
  int row  = pos_base + col;   // A: lane&15 = M row (position)

  // last sequence slot holds the freshly-written k (cache is input-only)
  const _Float16* krow = (row == SEQ - 1)
      ? (knew + (size_t)kv * HD)
      : (kc + ((size_t)kv * SEQ + row) * HD);

  // hoisted B fragments: lane=col picks its (zero-padded) q row; elements 0..7 are
  // chunk rows 8*half .. +8, 8..15 are chunk rows 16+8*half .. +8 -> two 16B loads
  const _Float16* qrow = qB + ((size_t)(kv * 16 + col)) * HD;
  v16h bfrag[4];
#pragma unroll
  for (int c4 = 0; c4 < 4; ++c4) {
    union { v16h v; v8h h[2]; } b;
    b.h[0] = *(const v8h*)(qrow + c4 * 32 + 8 * half);
    b.h[1] = *(const v8h*)(qrow + c4 * 32 + 16 + 8 * half);
    bfrag[c4] = b.v;
  }

  v8f acc;
#pragma unroll
  for (int e = 0; e < 8; ++e) acc[e] = 0.f;

#pragma unroll
  for (int c4 = 0; c4 < 4; ++c4) {
    int c = c4 * 32;
    // A operand: elements 0..7 = K[c+8*half .. +8), 8..15 = K[c+16+8*half .. +8)
    union { v16h v; v8h h[2]; } a;
    a.h[0] = *(const v8h*)(krow + c + 8 * half);
    a.h[1] = *(const v8h*)(krow + c + 16 + 8 * half);
    acc = __builtin_amdgcn_wmma_f32_16x16x32_f16(false, a.v, false, bfrag[c4],
                                                 (short)0, acc, false, false);
  }

  if (col < REP) {
    int qh = kv * REP + col;
#pragma unroll
    for (int g = 0; g < 8; ++g) {        // D: vgpr g -> M = g + 8*half
      int pos = pos_base + 8 * half + g;
      scores[(size_t)qh * SEQ + pos] = acc[g] * SCALEV + mask[pos];
    }
  }
}

// ---------------- softmax per head: p = exp(s - max) as fp16, sum saved ----------------
__global__ void softmax_kernel(const float* __restrict__ scores,
                               _Float16* __restrict__ p16,
                               float* __restrict__ rowsum) {
  __shared__ float red[256];
  int h = blockIdx.x, tid = threadIdx.x;
  const float* s = scores + (size_t)h * SEQ;
  float m = -1e30f;
  for (int i = tid; i < SEQ; i += 256) m = fmaxf(m, s[i]);
  red[tid] = m; __syncthreads();
  for (int k = 128; k > 0; k >>= 1) { if (tid < k) red[tid] = fmaxf(red[tid], red[tid + k]); __syncthreads(); }
  m = red[0]; __syncthreads();
  _Float16* p = p16 + (size_t)h * SEQ;
  float sum = 0.f;
  for (int i = tid; i < SEQ; i += 256) {
    float e = __expf(s[i] - m);
    p[i] = (_Float16)e;
    sum += e;
  }
  red[tid] = sum; __syncthreads();
  for (int k = 128; k > 0; k >>= 1) { if (tid < k) red[tid] += red[tid + k]; __syncthreads(); }
  if (tid == 0) rowsum[h] = red[0];
}

// ---------------- attn * V, split over sequence; V read once per 2 q heads ----------------
__global__ void attn_out_partial_kernel(const _Float16* __restrict__ vc,
                                        const _Float16* __restrict__ vnew,
                                        const _Float16* __restrict__ p16,
                                        float* __restrict__ opart) {
  int kv = blockIdx.y, s = blockIdx.x, t = threadIdx.x;   // block = 128 = HD
  int qh0 = kv * REP, qh1 = qh0 + 1;
  const int chunk = SEQ / SPLIT;                          // 256
  int start = s * chunk;
  __shared__ _Float16 p0s[128], p1s[128];
  float a0 = 0.f, a1 = 0.f;
  const _Float16* vbase = vc + (size_t)kv * SEQ * HD;
  for (int base = start; base < start + chunk; base += 128) {
    p0s[t] = p16[(size_t)qh0 * SEQ + base + t];
    p1s[t] = p16[(size_t)qh1 * SEQ + base + t];
    __syncthreads();
#pragma unroll 4
    for (int j = 0; j < 128; ++j) {
      float v = (float)vbase[(size_t)(base + j) * HD + t];
      a0 = fmaf((float)p0s[j], v, a0);
      a1 = fmaf((float)p1s[j], v, a1);
    }
    __syncthreads();
  }
  if (s == SPLIT - 1) {   // swap in the fresh v for the final position
    float d = (float)vnew[kv * HD + t] - (float)vbase[(size_t)(SEQ - 1) * HD + t];
    a0 = fmaf((float)p16[(size_t)qh0 * SEQ + SEQ - 1], d, a0);
    a1 = fmaf((float)p16[(size_t)qh1 * SEQ + SEQ - 1], d, a1);
  }
  opart[((size_t)qh0 * SPLIT + s) * HD + t] = a0;
  opart[((size_t)qh1 * SPLIT + s) * HD + t] = a1;
}

__global__ void attn_out_reduce_kernel(const float* __restrict__ opart,
                                       const float* __restrict__ rowsum,
                                       float* __restrict__ o) {
  int h = blockIdx.x, t = threadIdx.x;   // 16 x 128
  float a = 0.f;
  for (int s = 0; s < SPLIT; ++s) a += opart[((size_t)h * SPLIT + s) * HD + t];
  o[h * HD + t] = a / rowsum[h];
}

// ---------------- driver ----------------
extern "C" void kernel_launch(void* const* d_in, const int* in_sizes, int n_in,
                              void* d_out, int out_size, void* d_ws, size_t ws_size,
                              hipStream_t stream) {
  const float*    hidden  = (const float*)d_in[0];
  const float*    cosv    = (const float*)d_in[1];
  const float*    sinv    = (const float*)d_in[2];
  const float*    mask    = (const float*)d_in[3];
  const _Float16* kc      = (const _Float16*)d_in[4];
  const _Float16* vc      = (const _Float16*)d_in[5];
  const float*    w_in    = (const float*)d_in[6];
  const float*    wq      = (const float*)d_in[7];
  const float*    wk      = (const float*)d_in[8];
  const float*    wv      = (const float*)d_in[9];
  const float*    wo      = (const float*)d_in[10];
  const float*    w_post  = (const float*)d_in[11];
  const float*    wgate   = (const float*)d_in[12];
  const float*    wup     = (const float*)d_in[13];
  const float*    wdown   = (const float*)d_in[14];
  const float*    w_final = (const float*)d_in[15];
  const float*    wlm     = (const float*)d_in[16];

  // workspace carve-out (256B aligned slices)
  char* p = (char*)d_ws;
  auto alloc = [&](size_t bytes) -> char* {
    char* r = p; p += (bytes + 255) & ~(size_t)255; return r;
  };
  float* hnorm  = (float*)alloc(H * 4);
  float* qraw   = (float*)alloc(NQ * HD * 4);
  float* kraw   = (float*)alloc(NKV * HD * 4);
  float* vraw   = (float*)alloc(NKV * HD * 4);
  float* oattn  = (float*)alloc(NQ * HD * 4);
  float* h2     = (float*)alloc(H * 4);
  float* hn2    = (float*)alloc(H * 4);
  float* act    = (float*)alloc(INTER * 4);
  float* h3     = (float*)alloc(H * 4);
  float* last   = (float*)alloc(H * 4);
  float* rowsum = (float*)alloc(NQ * 4);
  float* scores = (float*)alloc((size_t)NQ * SEQ * 4);
  float* opart  = (float*)alloc((size_t)NQ * SPLIT * HD * 4);
  _Float16* qB     = (_Float16*)alloc((size_t)NKV * 16 * HD * 2);  // padded B operand
  _Float16* k16new = (_Float16*)alloc(NKV * HD * 2);
  _Float16* v16new = (_Float16*)alloc(NKV * HD * 2);
  _Float16* p16    = (_Float16*)alloc((size_t)NQ * SEQ * 2);

  // 1) input RMSNorm
  rmsnorm_kernel<<<1, 256, 0, stream>>>(hidden, w_in, hnorm);
  // 2) QKV projections (GEMV, wave-per-row)
  gemv_kernel<<<(NQ * HD) / 8, 256, 0, stream>>>(wq, hnorm, qraw, nullptr, H, NQ * HD);
  gemv_kernel<<<(NKV * HD) / 8, 256, 0, stream>>>(wk, hnorm, kraw, nullptr, H, NKV * HD);
  gemv_kernel<<<(NKV * HD) / 8, 256, 0, stream>>>(wv, hnorm, vraw, nullptr, H, NKV * HD);
  // 3) RoPE + fp16 staging (zero-padded WMMA B buffer + new k/v)
  {
    int total = NKV * 16 * HD + 2 * NKV * HD;   // 18432
    rope_stage_kernel<<<(total + 255) / 256, 256, 0, stream>>>(
        qraw, kraw, vraw, cosv, sinv, qB, k16new, v16new);
  }
  // 4) attention scores via WMMA f16 (K cache streamed, last slot redirected)
  attn_scores_kernel<<<dim3(SEQ / 128, NKV), 256, 0, stream>>>(kc, k16new, qB, mask, scores);
  // 5) softmax per head
  softmax_kernel<<<NQ, 256, 0, stream>>>(scores, p16, rowsum);
  // 6) attn * V (split-K) + reduce
  attn_out_partial_kernel<<<dim3(SPLIT, NKV), HD, 0, stream>>>(vc, v16new, p16, opart);
  attn_out_reduce_kernel<<<NQ, HD, 0, stream>>>(opart, rowsum, oattn);
  // 7) output projection + residual
  gemv_kernel<<<H / 8, 256, 0, stream>>>(wo, oattn, h2, hidden, NQ * HD, H);
  // 8) post-attention RMSNorm
  rmsnorm_kernel<<<1, 256, 0, stream>>>(h2, w_post, hn2);
  // 9) fused gate/up + SiLU
  gateup_kernel<<<INTER / 8, 256, 0, stream>>>(wgate, wup, hn2, act, H, INTER);
  // 10) down projection + residual
  gemv_kernel<<<H / 8, 256, 0, stream>>>(wdown, act, h3, h2, INTER, H);
  // 11) final RMSNorm
  rmsnorm_kernel<<<1, 256, 0, stream>>>(h3, w_final, last);
  // 12) LM head -> logits
  gemv_kernel<<<(VOCAB + 7) / 8, 256, 0, stream>>>(wlm, last, (float*)d_out, nullptr, H, VOCAB);
}
